// MultiheadLatentAttention_34308198760480
// MI455X (gfx1250) — matrix operand order repaired and checked by reading, not compiled, not run on hardware
//
#include <hip/hip_runtime.h>
#include <hip/hip_bf16.h>
#include <math.h>

// ---------------- problem constants ----------------
#define B_   2
#define S_   1024
#define L_   1024
#define DIMQ 2048
#define H_   16
#define KH   128      // per-head dim
#define DC   512      // compressed kv dim
#define DCQ  1536
#define R_   64       // retrieval dim
#define HK   2048     // H_*KH
#define BS   2048     // B_*S_

typedef __attribute__((ext_vector_type(16))) __bf16 v16bf;
typedef __attribute__((ext_vector_type(8)))  __bf16 v8bf;
typedef __attribute__((ext_vector_type(8)))  float  v8f;

// ---------------- helpers ----------------
__device__ __forceinline__ unsigned short f2bf(float f) {
  unsigned int u = __float_as_uint(f);
  u += 0x7fffu + ((u >> 16) & 1u);          // round-to-nearest-even
  return (unsigned short)(u >> 16);
}

__device__ __forceinline__ v16bf make_v16(const unsigned short* p0, const unsigned short* p1) {
  union { v16bf v; v8bf h[2]; } u;
  u.h[0] = *reinterpret_cast<const v8bf*>(p0);
  u.h[1] = *reinterpret_cast<const v8bf*>(p1);
  return u.v;
}

// A fragment (16x32 bf16, M x K), tile row-major, leading dim ld.
// ISA layout: lanes 0-15 hold M=lane, K=0..7 & 16..23; lanes 16-31 K=8..15 & 24..31.
__device__ __forceinline__ v16bf frag_a(const unsigned short* t, int ld, int row0, int k0, int lane) {
  const int m  = lane & 15;
  const int kb = (lane < 16) ? 0 : 8;
  const unsigned short* p = t + (size_t)(row0 + m) * ld + k0 + kb;
  return make_v16(p, p + 16);
}

// B fragment (32x16 bf16, K x N) from B^T tile Bt[n][k], leading dim ld.
// ISA layout: lanes 0-15 hold N=lane, K=0..7 & 16..23; lanes 16-31 K=8..15 & 24..31.
__device__ __forceinline__ v16bf frag_b(const unsigned short* t, int ld, int n0, int k0, int lane) {
  const int n  = lane & 15;
  const int kb = (lane < 16) ? 0 : 8;
  const unsigned short* p = t + (size_t)(n0 + n) * ld + k0 + kb;
  return make_v16(p, p + 16);
}

// ---- CDNA5 async global->LDS DMA (ASYNCcnt-tracked), VFLAT GVS form ----
__device__ __forceinline__ unsigned lds_addr(const void* p) {
  return (unsigned)(unsigned long long)p;   // LDS aperture: addr[31:0] is the LDS offset
}
// copy 16 contiguous bf16 (2 x b128); inst offset applies to LDS and global alike
__device__ __forceinline__ void async_row32(unsigned lds, unsigned voff, const void* base) {
  asm volatile("global_load_async_to_lds_b128 %0, %1, %2\n\t"
               "global_load_async_to_lds_b128 %0, %1, %2 offset:16"
               :: "v"(lds), "v"(voff), "s"(base) : "memory");
}
// copy 32 contiguous bf16 (4 x b128)
__device__ __forceinline__ void async_row64(unsigned lds, unsigned voff, const void* base) {
  asm volatile("global_load_async_to_lds_b128 %0, %1, %2\n\t"
               "global_load_async_to_lds_b128 %0, %1, %2 offset:16\n\t"
               "global_load_async_to_lds_b128 %0, %1, %2 offset:32\n\t"
               "global_load_async_to_lds_b128 %0, %1, %2 offset:48"
               :: "v"(lds), "v"(voff), "s"(base) : "memory");
}
__device__ __forceinline__ void wait_async0() {
  asm volatile("s_wait_asynccnt 0x0" ::: "memory");
}

// ---------------- elementwise kernels ----------------
__global__ void cvt_bf16_kernel(const float* __restrict__ src, unsigned short* __restrict__ dst, int n) {
  int i = blockIdx.x * 256 + threadIdx.x;
  if (i < n) dst[i] = f2bf(src[i]);
}

// transpose [rows][cols] f32 -> [cols][rows] bf16, batched along z
__global__ void transpose_cvt_kernel(const float* __restrict__ src, unsigned short* __restrict__ dst,
                                     int rows, int cols) {
  long long base = (long long)blockIdx.z * rows * cols;
  int i = blockIdx.x * 256 + threadIdx.x;
  if (i >= rows * cols) return;
  int r = i / cols, c = i % cols;
  dst[base + (long long)c * rows + r] = f2bf(src[base + i]);
}

// RoPE on compressed kv; writes [b][l][dc] and transposed [b][dc][l] (both bf16)
__global__ void rope_bf16_kernel(const float* __restrict__ kv, unsigned short* __restrict__ out,
                                 unsigned short* __restrict__ outT, int n) {
  int i = blockIdx.x * 256 + threadIdx.x;
  if (i >= n) return;
  int d  = i % DC;
  int bl = i / DC;
  int l  = bl % L_;
  int b  = bl / L_;
  int j  = d >> 1;
  float inv = __expf(-((float)(2 * j) / (float)DC) * 9.210340371976184f);  // ln(10000)
  float fr  = (float)l * inv;
  float sv  = __sinf(fr), cv = __cosf(fr);
  float x   = kv[i];
  float rot = (d < DC / 2) ? -kv[(long long)bl * DC + d + DC / 2]
                           :  kv[(long long)bl * DC + d - DC / 2];
  unsigned short v = f2bf(x * cv + rot * sv);
  out[i] = v;
  outT[((long long)b * DC + d) * L_ + l] = v;
}

// W_comb[h] = w_kc_q[h] @ W_qr[h], stored transposed [h][R][KH] (one-time, 0.4 GFLOP)
__global__ void combine_q_kernel(const float* __restrict__ w_kc_q, const float* __restrict__ W_qr,
                                 unsigned short* __restrict__ Wc) {
  int i = blockIdx.x * 256 + threadIdx.x;
  if (i >= H_ * KH * R_) return;
  int r = i % R_;
  int k = (i / R_) % KH;
  int h = i / (R_ * KH);
  const float* a = w_kc_q + ((long long)h * KH + k) * DCQ;
  const float* b = W_qr + (long long)h * DCQ * R_ + r;
  float acc = 0.f;
  for (int q = 0; q < DCQ; ++q) acc += a[q] * b[(long long)q * R_];
  Wc[((long long)h * R_ + r) * KH + k] = f2bf(acc);   // B^T layout
}

// ---------------- generic tiled bf16 WMMA GEMM (B given transposed) ----------------
// C[M,N] = A[M,Kd] x BT[N,Kd]^T, row-major bf16, fp32 accumulate.
// Requires M % 128 == 0, Kd % 32 == 0 (true for every call site); N may be partial.
// 128x128 block tile, BK=32, 8 wave32 waves in 4x2, wave tile 32x64 -> 8 accum frags.
// LDS double-buffered; tiles filled with GLOBAL_LOAD_ASYNC_TO_LDS_B128 overlapping WMMA.
__global__ __launch_bounds__(256)
void gemm_bf16_kernel(const unsigned short* __restrict__ A,
                      const unsigned short* __restrict__ BT,
                      void* __restrict__ Cp,
                      int M, int N, int Kd, int lda, int ldbt, int ldc,
                      int inner,
                      long long aOut, long long aIn, long long bOut, long long bIn,
                      long long cOut, long long cIn, int outBf16) {
  __shared__ __align__(16) unsigned short As[2][128 * 32];
  __shared__ __align__(16) unsigned short Bs[2][128 * 32];

  const int tid   = threadIdx.x;
  const int lane  = tid & 31;
  const int wave  = tid >> 5;
  const int waveM = wave >> 1;   // 0..3
  const int waveN = wave & 1;    // 0..1
  const int m0 = blockIdx.y * 128;
  const int n0 = blockIdx.x * 128;
  const int ob = blockIdx.z / inner, ib = blockIdx.z % inner;
  const unsigned short* Ab = A  + ob * aOut + ib * aIn;
  const unsigned short* Bb = BT + ob * bOut + ib * bIn;
  const long long cBase    = ob * cOut + ib * cIn;

  const int row  = tid >> 1;            // 0..127: tile row this thread stages
  const int part = (tid & 1) << 4;      // 0 / 16 halves within the 32-half row
  const int am   = m0 + row;                              // always < M
  const int bn   = (n0 + row < N) ? (n0 + row) : (N - 1); // branchless clamp

  v8f acc[2][4];
#pragma unroll
  for (int fm = 0; fm < 2; ++fm)
#pragma unroll
    for (int fn = 0; fn < 4; ++fn)
#pragma unroll
      for (int e = 0; e < 8; ++e) acc[fm][fn][e] = 0.f;

  const int nIter = Kd >> 5;
  // prologue: fill buffer 0
  async_row32(lds_addr(&As[0][row * 32 + part]),
              (unsigned)((((long long)am * lda) + part) * 2), Ab);
  async_row32(lds_addr(&Bs[0][row * 32 + part]),
              (unsigned)((((long long)bn * ldbt) + part) * 2), Bb);
  wait_async0();
  __syncthreads();

  for (int i = 0; i < nIter; ++i) {
    const int cur = i & 1;
    if (i + 1 < nIter) {   // prefetch next k-slice while WMMAs run on current
      const int k1 = (i + 1) << 5;
      async_row32(lds_addr(&As[cur ^ 1][row * 32 + part]),
                  (unsigned)((((long long)am * lda) + k1 + part) * 2), Ab);
      async_row32(lds_addr(&Bs[cur ^ 1][row * 32 + part]),
                  (unsigned)((((long long)bn * ldbt) + k1 + part) * 2), Bb);
    }
    v16bf afr[2];
#pragma unroll
    for (int fm = 0; fm < 2; ++fm) afr[fm] = frag_a(As[cur], 32, waveM * 32 + fm * 16, 0, lane);
#pragma unroll
    for (int fn = 0; fn < 4; ++fn) {
      v16bf bfr = frag_b(Bs[cur], 32, waveN * 64 + fn * 16, 0, lane);
#pragma unroll
      for (int fm = 0; fm < 2; ++fm)
        acc[fm][fn] = __builtin_amdgcn_wmma_f32_16x16x32_bf16(
            false, afr[fm], false, bfr, (short)0, acc[fm][fn], false, false);
    }
    if (i + 1 < nIter) wait_async0();
    __syncthreads();
  }

  const int rowHi = (lane < 16) ? 0 : 8;
  const int colL  = lane & 15;
#pragma unroll
  for (int fm = 0; fm < 2; ++fm)
#pragma unroll
    for (int fn = 0; fn < 4; ++fn) {
      int cb = n0 + waveN * 64 + fn * 16 + colL;
      if (cb >= N) continue;
      int rb = m0 + waveM * 32 + fm * 16 + rowHi;
#pragma unroll
      for (int v = 0; v < 8; ++v) {
        long long idx = cBase + (long long)(rb + v) * ldc + cb;
        if (outBf16) ((unsigned short*)Cp)[idx] = f2bf(acc[fm][fn][v]);
        else         ((float*)Cp)[idx]          = acc[fm][fn][v];
      }
    }
}

// ---------------- flash attention core ----------------
// grid: (DC/128, S/128, B*H).  Q tile [128,64]; per 128-L chunk: scores = Q K^T (WMMA)
// -> LDS, online softmax, ctx += P V (WMMA) over a 128-wide DC chunk. V^T layout so all
// tile fills are contiguous-row async DMAs.
__global__ __launch_bounds__(256)
void mla_flash_kernel(const unsigned short* __restrict__ qr,   // [B*H][S][R]
                      const unsigned short* __restrict__ kr,   // [B*H][L][R]
                      const unsigned short* __restrict__ kvT,  // [B][DC][L]
                      unsigned short* __restrict__ ctx) {      // [B*H][S][DC]
  extern __shared__ char smem[];
  unsigned short* Qs  = (unsigned short*)(smem);            // 128*64*2  = 16384  [s][r]
  unsigned short* Ks  = (unsigned short*)(smem + 16384);    // 128*64*2  = 16384  [l][r]
  float*          Sc  = (float*)(smem + 32768);             // 128*128*4 = 65536
  unsigned short* Ps  = (unsigned short*)(smem + 98304);    // 128*128*2 = 32768  attn bf16 [s][l]
  unsigned short* KVt = (unsigned short*)(smem + 131072);   // 128*32*2  = 8192   [dc][l]
  float* rowMax   = (float*)(smem + 139264);
  float* rowSum   = (float*)(smem + 139776);
  float* rowScale = (float*)(smem + 140288);                // total 140800 B

  const int tid   = threadIdx.x;
  const int lane  = tid & 31;
  const int wave  = tid >> 5;
  const int waveM = wave >> 1;
  const int waveN = wave & 1;
  const int dc0 = blockIdx.x * 128;
  const int s0  = blockIdx.y * 128;
  const int bh  = blockIdx.z;
  const int b   = bh >> 4;       // H_ == 16

  const unsigned short* qBase = qr  + (long long)bh * S_ * R_;
  const unsigned short* kBase = kr  + (long long)bh * L_ * R_;
  const unsigned short* vBase = kvT + (long long)b  * DC * L_;

  const int row  = tid >> 1;
  const int p64  = (tid & 1) << 5;   // 32-half part of a 64-half row
  const int p32  = (tid & 1) << 4;   // 16-half part of a 32-half row

  // Q tile [128][64] via async DMA (waited together with first K tile)
  async_row64(lds_addr(&Qs[row * 64 + p64]),
              (unsigned)((((long long)(s0 + row)) * R_ + p64) * 2), qBase);
  if (tid < 128) { rowMax[tid] = -3.0e38f; rowSum[tid] = 0.f; }

  v8f acc[2][4];
#pragma unroll
  for (int fm = 0; fm < 2; ++fm)
#pragma unroll
    for (int fn = 0; fn < 4; ++fn)
#pragma unroll
      for (int e = 0; e < 8; ++e) acc[fm][fn][e] = 0.f;

  const int rowHi = (lane < 16) ? 0 : 8;
  const int colL  = lane & 15;

  for (int l0 = 0; l0 < L_; l0 += 128) {
    // K tile [128][64] ([l][r] is already the B^T layout for the score GEMM)
    async_row64(lds_addr(&Ks[row * 64 + p64]),
                (unsigned)((((long long)(l0 + row)) * R_ + p64) * 2), kBase);
    wait_async0();
    __syncthreads();

    // scores tile [128 s][128 l]
    v8f sc[2][4];
#pragma unroll
    for (int fm = 0; fm < 2; ++fm)
#pragma unroll
      for (int fn = 0; fn < 4; ++fn)
#pragma unroll
        for (int e = 0; e < 8; ++e) sc[fm][fn][e] = 0.f;

#pragma unroll
    for (int k0 = 0; k0 < R_; k0 += 32) {
      v16bf afr[2];
#pragma unroll
      for (int fm = 0; fm < 2; ++fm) afr[fm] = frag_a(Qs, 64, waveM * 32 + fm * 16, k0, lane);
#pragma unroll
      for (int fn = 0; fn < 4; ++fn) {
        v16bf bfr = frag_b(Ks, 64, waveN * 64 + fn * 16, k0, lane);
#pragma unroll
        for (int fm = 0; fm < 2; ++fm)
          sc[fm][fn] = __builtin_amdgcn_wmma_f32_16x16x32_bf16(
              false, afr[fm], false, bfr, (short)0, sc[fm][fn], false, false);
      }
    }
    // spill scores to LDS
#pragma unroll
    for (int fm = 0; fm < 2; ++fm)
#pragma unroll
      for (int fn = 0; fn < 4; ++fn)
#pragma unroll
        for (int v = 0; v < 8; ++v)
          Sc[(waveM * 32 + fm * 16 + rowHi + v) * 128 + waveN * 64 + fn * 16 + colL] = sc[fm][fn][v];
    __syncthreads();

    // online softmax per row
    if (tid < 128) {
      const float* srow = &Sc[tid * 128];
      float om = rowMax[tid];
      float m = om;
      for (int c = 0; c < 128; ++c) m = fmaxf(m, srow[c]);
      float scal = __expf(om - m);
      float sum = 0.f;
      unsigned short* prow = &Ps[tid * 128];
      for (int c = 0; c < 128; ++c) {
        float e = __expf(srow[c] - m);
        sum += e;
        prow[c] = f2bf(e);
      }
      rowSum[tid] = rowSum[tid] * scal + sum;
      rowMax[tid] = m;
      rowScale[tid] = scal;
    }
    __syncthreads();

    // rescale running ctx accumulators
#pragma unroll
    for (int fm = 0; fm < 2; ++fm) {
#pragma unroll
      for (int v = 0; v < 8; ++v) {
        float f = rowScale[waveM * 32 + fm * 16 + rowHi + v];
#pragma unroll
        for (int fn = 0; fn < 4; ++fn) acc[fm][fn][v] *= f;
      }
    }

    // ctx += P[128 s x 128 l] @ V[128 l x 128 dc]   (V staged from kv^T, rows = dc)
#pragma unroll
    for (int k0 = 0; k0 < 128; k0 += 32) {
      async_row32(lds_addr(&KVt[row * 32 + p32]),
                  (unsigned)((((long long)(dc0 + row)) * L_ + l0 + k0 + p32) * 2), vBase);
      wait_async0();
      __syncthreads();
      v16bf afr[2];
#pragma unroll
      for (int fm = 0; fm < 2; ++fm) afr[fm] = frag_a(Ps, 128, waveM * 32 + fm * 16, k0, lane);
#pragma unroll
      for (int fn = 0; fn < 4; ++fn) {
        v16bf bfr = frag_b(KVt, 32, waveN * 64 + fn * 16, 0, lane);
#pragma unroll
        for (int fm = 0; fm < 2; ++fm)
          acc[fm][fn] = __builtin_amdgcn_wmma_f32_16x16x32_bf16(
              false, afr[fm], false, bfr, (short)0, acc[fm][fn], false, false);
      }
      __syncthreads();
    }
  }

  // normalize and store ctx_c (bf16)
#pragma unroll
  for (int fm = 0; fm < 2; ++fm)
#pragma unroll
    for (int v = 0; v < 8; ++v) {
      int srow = waveM * 32 + fm * 16 + rowHi + v;
      float invs = 1.0f / rowSum[srow];
#pragma unroll
      for (int fn = 0; fn < 4; ++fn) {
        long long idx = (long long)bh * S_ * DC + (long long)(s0 + srow) * DC +
                        dc0 + waveN * 64 + fn * 16 + colL;
        ctx[idx] = f2bf(acc[fm][fn][v] * invs);
      }
    }
}

// ---------------- workspace layout (bytes) ----------------
static const size_t OFF_KV   = 0;                          // kv_rope bf16   [B][L][DC]     2 MB
static const size_t OFF_HQ   = OFF_KV   + 2097152;         // hidden bf16    [BS][DIMQ]     8 MB
static const size_t OFF_WQ   = OFF_HQ   + 8388608;         // Wq bf16 (B^T)  [HK][DIMQ]     8 MB
static const size_t OFF_QHK  = OFF_WQ   + 8388608;         // q_hk bf16      [BS][HK]       8 MB
static const size_t OFF_WCMB = OFF_QHK  + 8388608;         // W_comb^T bf16  [H][R][KH]     256 KB
static const size_t OFF_QR   = OFF_WCMB + 262144;          // q_r bf16       [B*H][S][R]    4 MB
static const size_t OFF_WKR  = OFF_QR   + 4194304;         // W_kr^T bf16    [H][R][DC]     1 MB
static const size_t OFF_KR   = OFF_WKR  + 1048576;         // k_r bf16       [B*H][L][R]    4 MB
static const size_t OFF_CTXC = OFF_KR   + 4194304;         // ctx_c bf16     [B*H][S][DC]   32 MB
static const size_t OFF_WKV  = OFF_CTXC + 33554432;        // w_kc_kv bf16   [H][KH][DC]    2 MB
static const size_t OFF_CTXL = OFF_WKV  + 2097152;         // ctx_lat bf16   [BS][HK]       8 MB
static const size_t OFF_WO   = OFF_CTXL + 8388608;         // Wo bf16 (B^T)  [DIMQ][HK]     8 MB
static const size_t OFF_KVT  = OFF_WO   + 8388608;         // kv_rope^T bf16 [B][DC][L]     2 MB

static void launch_gemm(const unsigned short* A, const unsigned short* BT, void* C,
                        int M, int N, int Kd, int lda, int ldbt, int ldc,
                        int batch, int inner,
                        long long aOut, long long aIn, long long bOut, long long bIn,
                        long long cOut, long long cIn, int outBf16, hipStream_t s) {
  dim3 g((N + 127) / 128, (M + 127) / 128, batch);
  gemm_bf16_kernel<<<g, 256, 0, s>>>(A, BT, C, M, N, Kd, lda, ldbt, ldc, inner,
                                     aOut, aIn, bOut, bIn, cOut, cIn, outBf16);
}

extern "C" void kernel_launch(void* const* d_in, const int* in_sizes, int n_in,
                              void* d_out, int out_size, void* d_ws, size_t ws_size,
                              hipStream_t stream) {
  (void)in_sizes; (void)n_in; (void)out_size; (void)ws_size;
  const float* hidden_q = (const float*)d_in[0];
  const float* kv_c     = (const float*)d_in[1];
  const float* Wq       = (const float*)d_in[2];
  const float* w_kc_q   = (const float*)d_in[3];
  const float* w_kc_kv  = (const float*)d_in[4];
  const float* W_qr     = (const float*)d_in[5];
  const float* W_kr     = (const float*)d_in[6];
  const float* Wo       = (const float*)d_in[7];

  char* ws = (char*)d_ws;
  unsigned short* kvbf  = (unsigned short*)(ws + OFF_KV);
  unsigned short* hqbf  = (unsigned short*)(ws + OFF_HQ);
  unsigned short* wqbf  = (unsigned short*)(ws + OFF_WQ);
  unsigned short* qhk   = (unsigned short*)(ws + OFF_QHK);
  unsigned short* wcmb  = (unsigned short*)(ws + OFF_WCMB);
  unsigned short* qrb   = (unsigned short*)(ws + OFF_QR);
  unsigned short* wkrT  = (unsigned short*)(ws + OFF_WKR);
  unsigned short* krb   = (unsigned short*)(ws + OFF_KR);
  unsigned short* ctxc  = (unsigned short*)(ws + OFF_CTXC);
  unsigned short* wkvbf = (unsigned short*)(ws + OFF_WKV);
  unsigned short* ctxl  = (unsigned short*)(ws + OFF_CTXL);
  unsigned short* wobf  = (unsigned short*)(ws + OFF_WO);
  unsigned short* kvTbf = (unsigned short*)(ws + OFF_KVT);

  // --- precision / layout preparation (Wq, Wo, w_kc_kv are already B^T) ---
  { int n = BS * DIMQ;    cvt_bf16_kernel<<<(n + 255) / 256, 256, 0, stream>>>(hidden_q, hqbf, n); }
  { int n = HK * DIMQ;    cvt_bf16_kernel<<<(n + 255) / 256, 256, 0, stream>>>(Wq, wqbf, n); }
  { int n = DIMQ * HK;    cvt_bf16_kernel<<<(n + 255) / 256, 256, 0, stream>>>(Wo, wobf, n); }
  { int n = H_ * KH * DC; cvt_bf16_kernel<<<(n + 255) / 256, 256, 0, stream>>>(w_kc_kv, wkvbf, n); }
  { int n = B_ * L_ * DC; rope_bf16_kernel<<<(n + 255) / 256, 256, 0, stream>>>(kv_c, kvbf, kvTbf, n); }
  transpose_cvt_kernel<<<dim3((DC * R_ + 255) / 256, 1, H_), 256, 0, stream>>>(W_kr, wkrT, DC, R_);
  { int n = H_ * KH * R_; combine_q_kernel<<<(n + 255) / 256, 256, 0, stream>>>(w_kc_q, W_qr, wcmb); }

  // --- q_hk = hidden @ Wq^T : [2048,2048]x[2048,2048] ---
  launch_gemm(hqbf, wqbf, qhk, BS, HK, DIMQ, DIMQ, DIMQ, HK,
              1, 1, 0, 0, 0, 0, 0, 0, 1, stream);

  // --- q_r[b,h] = q_hk[:,h] @ W_comb[h] : per (b,h) [1024,128]x[128,64] ---
  launch_gemm(qhk, wcmb, qrb, S_, R_, KH, HK, KH, R_,
              B_ * H_, H_,
              (long long)S_ * HK, KH,
              0, (long long)R_ * KH,
              (long long)H_ * S_ * R_, (long long)S_ * R_, 1, stream);

  // --- k_r[b,h] = kv_rope[b] @ W_kr[h] : per (b,h) [1024,512]x[512,64] ---
  launch_gemm(kvbf, wkrT, krb, L_, R_, DC, DC, DC, R_,
              B_ * H_, H_,
              (long long)L_ * DC, 0,
              0, (long long)R_ * DC,
              (long long)H_ * L_ * R_, (long long)L_ * R_, 1, stream);

  // --- flash attention: ctx_c = softmax(q_r k_r^T) @ kv_rope ---
  mla_flash_kernel<<<dim3(DC / 128, S_ / 128, B_ * H_), 256, 140800, stream>>>(qrb, krb, kvTbf, ctxc);

  // --- ctx_lat[b,:,h,:] = ctx_c[b,h] @ w_kc_kv[h]^T : per (b,h) [1024,512]x[512,128] ---
  launch_gemm(ctxc, wkvbf, ctxl, S_, KH, DC, DC, DC, HK,
              B_ * H_, H_,
              (long long)H_ * S_ * DC, (long long)S_ * DC,
              0, (long long)KH * DC,
              (long long)S_ * HK, KH, 1, stream);

  // --- out = ctx_lat @ Wo^T : [2048,2048]x[2048,2048] -> fp32 d_out ---
  launch_gemm(ctxl, wobf, d_out, BS, DIMQ, HK, HK, HK, DIMQ,
              1, 1, 0, 0, 0, 0, 0, 0, 0, stream);
}
